// Encoder_30949534335025
// MI455X (gfx1250) — matrix operand order repaired and checked by reading, not compiled, not run on hardware
//
#include <hip/hip_runtime.h>
#include <hip/hip_bf16.h>

// ---------------------------------------------------------------------------
// Problem constants (match reference)
// ---------------------------------------------------------------------------
#define NN      4096      // graph nodes
#define F_IN    512
#define HH      512
#define NH      4
#define DH      128
#define BB      1024
#define NCTX    4
#define LL      64
#define BC      (BB * NCTX)   // 4096 sequences
#define H3      (3 * HH)      // 1536

#define XROW    260           // padded dword row stride for GRU x/h tiles

typedef __bf16 bf16_t;
typedef __attribute__((ext_vector_type(16))) __bf16        v16bf;
typedef __attribute__((ext_vector_type(2)))  __bf16        v2bf;
typedef __attribute__((ext_vector_type(8)))  float         v8f;

#define WMMA_BF16(A_, B_, C_) \
    __builtin_amdgcn_wmma_f32_16x16x32_bf16(false, (A_), false, (B_), (short)0, (C_), false, false)

// CDNA5 async memory->LDS staging (ASYNCcnt-tracked), guarded so the file
// still compiles (with synchronous fallback) if the builtins are absent.
// Probe-confirmed signature: (v4i AS1* gsrc, v4i AS3* lds_dst, imm offset, imm cpol)
#if __has_builtin(__builtin_amdgcn_global_load_async_to_lds_b128) && \
    __has_builtin(__builtin_amdgcn_global_load_async_to_lds_b32)  && \
    __has_builtin(__builtin_amdgcn_s_wait_asynccnt)
#define USE_ASYNC 1
typedef int v4i_t __attribute__((__vector_size__(16)));
#define GAS128(p) ((__attribute__((address_space(1))) v4i_t*)(p))
#define LAS128(p) ((__attribute__((address_space(3))) v4i_t*)(p))
#define GAS32(p)  ((__attribute__((address_space(1))) int*)(p))
#define LAS32(p)  ((__attribute__((address_space(3))) int*)(p))
#else
#define USE_ASYNC 0
#endif

__device__ inline unsigned int pack_bf16(float lo, float hi) {
    v2bf p;
    p[0] = (bf16_t)lo;
    p[1] = (bf16_t)hi;
    return __builtin_bit_cast(unsigned int, p);
}

// 32-byte fragment assembled from two b128 loads, bit-cast to v16bf.
struct U8x4 { uint4 lo, hi; };

__device__ inline v16bf frag_from(const unsigned int* __restrict__ p, int skip_dw) {
    U8x4 u;
    u.lo = *(const uint4*)p;
    u.hi = *(const uint4*)(p + skip_dw);
    return __builtin_bit_cast(v16bf, u);
}

// ---------------------------------------------------------------------------
// WMMA fragment loaders (contiguous k = 16*half + s order, wave32), b128-granular:
//   A 16x32 bf16 from row-major dword tile (ldad dwords/row):
//     lane -> row (lane&15); dwords = half*8 + [0..7] contiguous
//   B 32x16 bf16 from kp-contiguous tile: lane col = lane&15;
//     dwords kp = 8*half + [0..7] contiguous
// ---------------------------------------------------------------------------
__device__ inline v16bf load_a_frag(const unsigned int* __restrict__ rowd,
                                    int ldad, int lane) {
    const int half = lane >> 4;
    const int m    = lane & 15;
    return frag_from(rowd + (size_t)m * ldad + half * 8, 4);
}

// Bsw tile: [col][20] dwords, col-major fragment run
__device__ inline v16bf load_b_frag_lds(const unsigned int* __restrict__ Bsw,
                                        int lane) {
    const int col = lane & 15;
    return frag_from(Bsw + (size_t)col * 20 + (lane >> 4) * 8, 4);
}

// Wsw global block: base -> ((kpb*H3 + j0) * 16); per-col 16 contiguous dwords
__device__ inline v16bf load_b_frag_glb(const unsigned int* __restrict__ Wsw,
                                        int lane) {
    return frag_from(Wsw + (size_t)(lane & 15) * 16 + (lane >> 4) * 8, 4);
}

// ---------------------------------------------------------------------------
// Generic bf16 WMMA GEMM: C[M,N] (f32, +bias) = A[M,K] * Bp, row-major A,
// B pre-packed as Bp[K/2][N] dwords. Block = 128 thr (4 waves), 64x64 tile,
// K-step 32, double-buffered (async-to-LDS staging when available).
// Requires M%64==0, N%64==0, K%32==0.
// ---------------------------------------------------------------------------
__global__ __launch_bounds__(128) void gemm_bf16_kernel(
    const bf16_t* __restrict__ A, const unsigned int* __restrict__ Bp,
    const float* __restrict__ bias, float* __restrict__ C,
    int M, int Nc, int K,
    long long Aoffz, long long Boffz, long long Coffz)
{
    A  += (long long)blockIdx.z * Aoffz;
    Bp += (long long)blockIdx.z * Boffz;
    C  += (long long)blockIdx.z * Coffz;

    const int tid  = threadIdx.x;
    const int wave = tid >> 5;
    const int lane = tid & 31;
    const int m0   = blockIdx.x * 64;
    const int n0   = blockIdx.y * 64;

    __shared__ unsigned int As[2][64][20];    // 64 rows x 16 dwords (+4 pad)
    __shared__ unsigned int Bsw[2][64][20];   // 64 cols x 16 kp dwords (+4 pad)

    v8f acc[4] = {};

    // stage one 64x32 A tile + transposed B tile into buffer b
    auto stage = [&](int b, int kks) {
        const int kp0 = kks >> 1;
#if USE_ASYNC
        for (int i = tid; i < 64 * 4; i += 128) {           // A: b128 async
            int r = i >> 2, q = i & 3;
            const unsigned int* g =
                (const unsigned int*)(A + (size_t)(m0 + r) * K + kks) + q * 4;
            __builtin_amdgcn_global_load_async_to_lds_b128(
                GAS128(g), LAS128(&As[b][r][q * 4]), 0, 0);
        }
        for (int i = tid; i < 64 * 16; i += 128) {          // B: b32 transpose scatter
            int c = i >> 4, kq = i & 15;
            const unsigned int* g = Bp + (size_t)(kp0 + kq) * Nc + n0 + c;
            __builtin_amdgcn_global_load_async_to_lds_b32(
                GAS32(g), LAS32(&Bsw[b][c][kq]), 0, 0);
        }
#else
        for (int i = tid; i < 64 * 4; i += 128) {
            int r = i >> 2, q = i & 3;
            ((uint4*)&As[b][r][0])[q] =
                ((const uint4*)(A + (size_t)(m0 + r) * K + kks))[q];
        }
        for (int i = tid; i < 64 * 4; i += 128) {
            int c = i >> 2, q = (i & 3) * 4;
            uint4 v;
            v.x = Bp[(size_t)(kp0 + q + 0) * Nc + n0 + c];
            v.y = Bp[(size_t)(kp0 + q + 1) * Nc + n0 + c];
            v.z = Bp[(size_t)(kp0 + q + 2) * Nc + n0 + c];
            v.w = Bp[(size_t)(kp0 + q + 3) * Nc + n0 + c];
            ((uint4*)&Bsw[b][c][0])[q >> 2] = v;
        }
#endif
    };

    stage(0, 0);
    int buf = 0;
    for (int kk = 0; kk < K; kk += 32) {
#if USE_ASYNC
        __builtin_amdgcn_s_wait_asynccnt(0);
#endif
        __syncthreads();
        if (kk + 32 < K) stage(buf ^ 1, kk + 32);           // prefetch next tile

        v16bf af = load_a_frag(&As[buf][wave * 16][0], 20, lane);
#pragma unroll
        for (int nt = 0; nt < 4; ++nt) {
            v16bf bf_ = load_b_frag_lds(&Bsw[buf][nt * 16][0], lane);
            acc[nt] = WMMA_BF16(af, bf_, acc[nt]);
        }
        buf ^= 1;
    }

    const int col = lane & 15;
    const int rh  = (lane >> 4) * 8;
#pragma unroll
    for (int nt = 0; nt < 4; ++nt) {
        int n = n0 + nt * 16 + col;
        float bv = bias ? bias[n] : 0.f;
#pragma unroll
        for (int e = 0; e < 8; ++e) {
            int m = m0 + wave * 16 + rh + e;
            C[(size_t)m * Nc + n] = acc[nt][e] + bv;
        }
    }
}

// ---------------------------------------------------------------------------
// Conversion / packing kernels
// ---------------------------------------------------------------------------
__global__ void gather_cast_kernel(const int* __restrict__ nodes,
                                   const float* __restrict__ emb,
                                   unsigned int* __restrict__ out)  // [NN][F_IN/2]
{
    int nrow = blockIdx.x;
    int src  = nodes[nrow];
    const float2* s = (const float2*)(emb + (size_t)src * F_IN);
    for (int c = threadIdx.x; c < F_IN / 2; c += blockDim.x) {
        float2 f = s[c];
        out[(size_t)nrow * (F_IN / 2) + c] = pack_bf16(f.x, f.y);
    }
}

// B row-major [K][N] f32 -> Bp[K/2][N] kp-packed dwords
__global__ void pack_b_kernel(const float* __restrict__ B, unsigned int* __restrict__ Bp,
                              int K, int Nc)
{
    int i = blockIdx.x * 256 + threadIdx.x;
    int total = (K / 2) * Nc;
    if (i < total) {
        int kp = i / Nc, n = i - kp * Nc;
        Bp[i] = pack_bf16(B[(size_t)(2 * kp) * Nc + n],
                          B[(size_t)(2 * kp + 1) * Nc + n]);
    }
}

// GRU weight: W[H3][HH] f32 -> fragment-swizzled Wsw[kpb][j][ki] dwords
// (kp = kpb*16 + ki;  Wsw dword = {W[j][2kp], W[j][2kp+1]})
__global__ void pack_gruw_kernel(const float* __restrict__ W, unsigned int* __restrict__ Wsw)
{
    int i = blockIdx.x * 256 + threadIdx.x;
    if (i < (HH / 2) * H3) {
        int kp = i / H3, j = i - kp * H3;
        const float* s = W + (size_t)j * HH + 2 * kp;
        int kpb = kp >> 4, ki = kp & 15;
        Wsw[((size_t)kpb * H3 + j) * 16 + ki] = pack_bf16(s[0], s[1]);
    }
}

// ---------------------------------------------------------------------------
// GAT: f1/f2 reductions   f1[h][n] = dot(hmat[h][n][:], a1[h][:])
// ---------------------------------------------------------------------------
__global__ __launch_bounds__(128) void f1f2_kernel(
    const float* __restrict__ hmat, const float* __restrict__ a1,
    const float* __restrict__ a2, float* __restrict__ f1, float* __restrict__ f2)
{
    int nidx = blockIdx.x, head = blockIdx.y, d = threadIdx.x;
    const float* row = hmat + ((size_t)head * NN + nidx) * DH;
    float v = row[d];
    __shared__ float r1[128], r2[128];
    r1[d] = v * a1[head * DH + d];
    r2[d] = v * a2[head * DH + d];
    __syncthreads();
    for (int s = 64; s > 0; s >>= 1) {
        if (d < s) { r1[d] += r1[d + s]; r2[d] += r2[d + s]; }
        __syncthreads();
    }
    if (d == 0) {
        f1[head * NN + nidx] = r1[0];
        f2[head * NN + nidx] = r2[0];
    }
}

// ---------------------------------------------------------------------------
// GAT fused attention row: e = leakyrelu(f1[n]+f2[m]) masked by adj,
// softmax over m, emb[n][head*DH + d] = sum_m att * hmat[head][m][d]
// block = 128 threads, grid = (NN, NH)
// ---------------------------------------------------------------------------
__global__ __launch_bounds__(128) void gat_att_row_kernel(
    const float* __restrict__ hmat, const int* __restrict__ adj,
    const float* __restrict__ f1, const float* __restrict__ f2,
    float* __restrict__ emb)   // [NN][NH*DH]
{
    const int nrow = blockIdx.x, head = blockIdx.y, tid = threadIdx.x;
    const float f1n = f1[head * NN + nrow];
    const int* arow = adj + (size_t)nrow * NN;
    const float* f2h = f2 + head * NN;

    __shared__ float red[128];

    float mx = -3.0e38f;
    for (int j = tid; j < NN; j += 128) {
        float e = f1n + f2h[j];
        e = e > 0.f ? e : 0.2f * e;
        e = arow[j] > 0 ? e : -9.0e15f;
        mx = fmaxf(mx, e);
    }
    red[tid] = mx; __syncthreads();
    for (int s = 64; s > 0; s >>= 1) {
        if (tid < s) red[tid] = fmaxf(red[tid], red[tid + s]);
        __syncthreads();
    }
    mx = red[0]; __syncthreads();

    float sum = 0.f;
    for (int j = tid; j < NN; j += 128) {
        float e = f1n + f2h[j];
        e = e > 0.f ? e : 0.2f * e;
        e = arow[j] > 0 ? e : -9.0e15f;
        sum += __expf(e - mx);
    }
    red[tid] = sum; __syncthreads();
    for (int s = 64; s > 0; s >>= 1) {
        if (tid < s) red[tid] += red[tid + s];
        __syncthreads();
    }
    const float inv = 1.0f / red[0];
    __syncthreads();

    const float* hh = hmat + (size_t)head * NN * DH;
    __shared__ float ps[128];
    float acc = 0.f;
    for (int c0 = 0; c0 < NN; c0 += 128) {
        int j = c0 + tid;
        float e = f1n + f2h[j];
        e = e > 0.f ? e : 0.2f * e;
        e = arow[j] > 0 ? e : -9.0e15f;
        ps[tid] = __expf(e - mx) * inv;
        __syncthreads();
        for (int i = 0; i < 128; ++i)
            acc += ps[i] * hh[(size_t)(c0 + i) * DH + tid];
        __syncthreads();
    }
    emb[(size_t)nrow * (NH * DH) + head * DH + tid] = acc;
}

// ---------------------------------------------------------------------------
// Persistent WMMA GRU: block owns 16 batch rows, loops all 64 timesteps.
// 4 waves; wave w owns hidden columns [w*128, w*128+128) (8 tiles of 16).
// r/z accumulate x@WiT + h@WhT in one accumulator; n/hn kept separate.
// Update masked by t < len (pack_padded last-hidden). grid = BC/16.
// Weights fragment-swizzled: Wsw[kpb][j][ki] dwords (2x b128 per fragment).
// ---------------------------------------------------------------------------
__global__ __launch_bounds__(128) void gru_kernel(
    const int* __restrict__ ctx,         // [BC][LL]
    const int* __restrict__ len,         // [BC]
    const float* __restrict__ word_emb,  // [WORD_VOCAB][HH]
    const unsigned int* __restrict__ WiP,// swizzled [HH/32][H3][16]
    const unsigned int* __restrict__ WhP,// swizzled
    const float* __restrict__ bi, const float* __restrict__ bh,
    float* __restrict__ outh)            // [BC][HH]
{
    const int tid  = threadIdx.x;
    const int wave = tid >> 5;
    const int lane = tid & 31;
    const int r0   = blockIdx.x * 16;

    __shared__ unsigned int xs[16][XROW];   // 16 rows x 256 dwords (+4 pad)
    __shared__ unsigned int hsh[16][XROW];
    __shared__ int lens[16];
    __shared__ int widx[16];

    for (int i = tid; i < 16 * XROW; i += 128) (&hsh[0][0])[i] = 0u;
    if (tid < 16) lens[tid] = len[r0 + tid];

    v8f hreg[8] = {};                    // f32 hidden, this wave's columns
    const int cb  = wave * 128;
    const int col = lane & 15;
    const int rh  = (lane >> 4) * 8;
    __syncthreads();

    for (int t = 0; t < LL; ++t) {
        if (tid < 16) widx[tid] = ctx[(size_t)(r0 + tid) * LL + t];
        __syncthreads();
        for (int i = tid; i < 16 * 256; i += 128) {
            int r = i >> 8, c = i & 255;
            float2 f = ((const float2*)(word_emb + (size_t)widx[r] * HH))[c];
            xs[r][c] = pack_bf16(f.x, f.y);
        }
        __syncthreads();

        v8f hnew[8];
#pragma unroll 1
        for (int nt = 0; nt < 8; ++nt) {
            const int jc = cb + nt * 16;
            v8f ar{}, az{}, an{}, ahn{};
            // gi: x @ WiT (r, z, n columns)
            for (int kk = 0; kk < HH; kk += 32) {
                v16bf af = load_a_frag(&xs[0][kk >> 1], XROW, lane);
                const unsigned int* wb = WiP + ((size_t)(kk >> 5) * H3 + jc) * 16;
                v16bf b0 = load_b_frag_glb(wb,                       lane);
                ar = WMMA_BF16(af, b0, ar);
                v16bf b1 = load_b_frag_glb(wb + (size_t)HH * 16,     lane);
                az = WMMA_BF16(af, b1, az);
                v16bf b2 = load_b_frag_glb(wb + (size_t)2 * HH * 16, lane);
                an = WMMA_BF16(af, b2, an);
            }
            // gh: h @ WhT (r, z share accumulators; hn separate)
            for (int kk = 0; kk < HH; kk += 32) {
                v16bf af = load_a_frag(&hsh[0][kk >> 1], XROW, lane);
                const unsigned int* wb = WhP + ((size_t)(kk >> 5) * H3 + jc) * 16;
                v16bf b0 = load_b_frag_glb(wb,                       lane);
                ar = WMMA_BF16(af, b0, ar);
                v16bf b1 = load_b_frag_glb(wb + (size_t)HH * 16,     lane);
                az = WMMA_BF16(af, b1, az);
                v16bf b2 = load_b_frag_glb(wb + (size_t)2 * HH * 16, lane);
                ahn = WMMA_BF16(af, b2, ahn);
            }
            const int j     = jc + col;
            const float br  = bi[j] + bh[j];
            const float bz  = bi[j + HH] + bh[j + HH];
            const float bin = bi[j + 2 * HH];
            const float bhn = bh[j + 2 * HH];
#pragma unroll
            for (int e = 0; e < 8; ++e) {
                const int m = rh + e;
                float r_ = 1.f / (1.f + __expf(-(ar[e] + br)));
                float z_ = 1.f / (1.f + __expf(-(az[e] + bz)));
                float nn = tanhf(an[e] + bin + r_ * (ahn[e] + bhn));
                float hp = hreg[nt][e];
                float h2 = (1.f - z_) * nn + z_ * hp;
                hnew[nt][e] = (t < lens[m]) ? h2 : hp;
            }
        }
        __syncthreads();   // all waves done reading hsh this step
#pragma unroll
        for (int nt = 0; nt < 8; ++nt) {
            hreg[nt] = hnew[nt];
            const int j = cb + nt * 16 + col;
#pragma unroll
            for (int e = 0; e < 8; ++e)
                ((bf16_t*)&hsh[0][0])[(size_t)(rh + e) * (2 * XROW) + j] = (bf16_t)hnew[nt][e];
        }
        __syncthreads();
    }

#pragma unroll
    for (int nt = 0; nt < 8; ++nt) {
        const int j = cb + nt * 16 + col;
#pragma unroll
        for (int e = 0; e < 8; ++e)
            outh[(size_t)(r0 + rh + e) * HH + j] = hreg[nt][e];
    }
}

// ---------------------------------------------------------------------------
// Build head/tail bf16 matrices [BB][2H] (stored packed as dwords):
//  cols 0..511  = elu(emb_in[map] + emb_out[map])
//  cols 512..   = mean over NCTX of GRU hidden (head<-out_h, tail<-in_h)
// ---------------------------------------------------------------------------
__global__ void build_headtail_kernel(
    const float* __restrict__ emb_in, const float* __restrict__ emb_out,
    const int* __restrict__ in_hs, const int* __restrict__ in_ts,
    const int* __restrict__ out_hs, const int* __restrict__ out_ts,
    const float* __restrict__ in_h, const float* __restrict__ out_h,
    unsigned int* __restrict__ headb, unsigned int* __restrict__ tailb) // [BB][HH] dwords
{
    const int b = blockIdx.x;
    const int ih = in_hs[b], oh = out_hs[b], it = in_ts[b], ot = out_ts[b];
    for (int cp = threadIdx.x; cp < HH / 2; cp += blockDim.x) {
        int c0 = 2 * cp;
        float g[2], gt[2], th[2], tt[2];
#pragma unroll
        for (int u = 0; u < 2; ++u) {
            int c = c0 + u;
            float v = emb_in[(size_t)ih * HH + c] + emb_out[(size_t)oh * HH + c];
            g[u] = v > 0.f ? v : (__expf(v) - 1.f);            // elu
            float w = emb_in[(size_t)it * HH + c] + emb_out[(size_t)ot * HH + c];
            gt[u] = w > 0.f ? w : (__expf(w) - 1.f);
            th[u] = 0.25f * (out_h[(size_t)(b * 4 + 0) * HH + c] +
                             out_h[(size_t)(b * 4 + 1) * HH + c] +
                             out_h[(size_t)(b * 4 + 2) * HH + c] +
                             out_h[(size_t)(b * 4 + 3) * HH + c]);
            tt[u] = 0.25f * (in_h[(size_t)(b * 4 + 0) * HH + c] +
                             in_h[(size_t)(b * 4 + 1) * HH + c] +
                             in_h[(size_t)(b * 4 + 2) * HH + c] +
                             in_h[(size_t)(b * 4 + 3) * HH + c]);
        }
        headb[(size_t)b * HH + cp]          = pack_bf16(g[0], g[1]);
        tailb[(size_t)b * HH + cp]          = pack_bf16(gt[0], gt[1]);
        headb[(size_t)b * HH + HH / 2 + cp] = pack_bf16(th[0], th[1]);
        tailb[(size_t)b * HH + HH / 2 + cp] = pack_bf16(tt[0], tt[1]);
    }
}

// ---------------------------------------------------------------------------
// Host launcher
// ---------------------------------------------------------------------------
static inline char* carve(char*& p, size_t bytes) {
    char* r = p;
    p += (bytes + 255) & ~(size_t)255;
    return r;
}

extern "C" void kernel_launch(void* const* d_in, const int* in_sizes, int n_in,
                              void* d_out, int out_size, void* d_ws, size_t ws_size,
                              hipStream_t stream)
{
    (void)in_sizes; (void)n_in; (void)out_size; (void)ws_size;

    const int*   in_nodes  = (const int*)d_in[0];
    const int*   out_nodes = (const int*)d_in[1];
    const int*   inner_adj = (const int*)d_in[2];
    const int*   outer_adj = (const int*)d_in[3];
    const int*   in_map_hs = (const int*)d_in[4];
    const int*   in_map_ts = (const int*)d_in[5];
    const int*   out_map_hs= (const int*)d_in[6];
    const int*   out_map_ts= (const int*)d_in[7];
    const int*   in_ctx    = (const int*)d_in[8];
    const int*   in_len    = (const int*)d_in[9];
    const int*   out_ctx   = (const int*)d_in[10];
    const int*   out_len   = (const int*)d_in[11];
    const float* node_emb  = (const float*)d_in[14];
    const float* word_emb  = (const float*)d_in[15];
    const float* W_att     = (const float*)d_in[16];
    const float* a1        = (const float*)d_in[17];
    const float* a2        = (const float*)d_in[18];
    const float* gin_Wi    = (const float*)d_in[19];
    const float* gin_Wh    = (const float*)d_in[20];
    const float* gin_bi    = (const float*)d_in[21];
    const float* gin_bh    = (const float*)d_in[22];
    const float* gout_Wi   = (const float*)d_in[23];
    const float* gout_Wh   = (const float*)d_in[24];
    const float* gout_bi   = (const float*)d_in[25];
    const float* gout_bh   = (const float*)d_in[26];
    const float* link_W    = (const float*)d_in[27];
    const float* link_b    = (const float*)d_in[28];
    const float* gen_W     = (const float*)d_in[29];
    const float* gen_b     = (const float*)d_in[30];
    float* out = (float*)d_out;

    // ---- workspace carving
    char* p = (char*)d_ws;
    bf16_t*       xin_bf   = (bf16_t*)carve(p, (size_t)NN * F_IN * 2);
    bf16_t*       xout_bf  = (bf16_t*)carve(p, (size_t)NN * F_IN * 2);
    unsigned int* watt_pk  = (unsigned int*)carve(p, (size_t)NH * (F_IN / 2) * DH * 4);
    float*        hin      = (float*)carve(p, (size_t)NH * NN * DH * 4);
    float*        hout     = (float*)carve(p, (size_t)NH * NN * DH * 4);
    float*        f1in     = (float*)carve(p, (size_t)NH * NN * 4);
    float*        f2in     = (float*)carve(p, (size_t)NH * NN * 4);
    float*        f1out    = (float*)carve(p, (size_t)NH * NN * 4);
    float*        f2out    = (float*)carve(p, (size_t)NH * NN * 4);
    float*        emb_in   = (float*)carve(p, (size_t)NN * (NH * DH) * 4);
    float*        emb_out  = (float*)carve(p, (size_t)NN * (NH * DH) * 4);
    unsigned int* WiP_in   = (unsigned int*)carve(p, (size_t)(HH / 2) * H3 * 4);
    unsigned int* WhP_in   = (unsigned int*)carve(p, (size_t)(HH / 2) * H3 * 4);
    unsigned int* WiP_out  = (unsigned int*)carve(p, (size_t)(HH / 2) * H3 * 4);
    unsigned int* WhP_out  = (unsigned int*)carve(p, (size_t)(HH / 2) * H3 * 4);
    float*        inh_seq  = (float*)carve(p, (size_t)BC * HH * 4);
    float*        outh_seq = (float*)carve(p, (size_t)BC * HH * 4);
    unsigned int* headb    = (unsigned int*)carve(p, (size_t)BB * HH * 4);
    unsigned int* tailb    = (unsigned int*)carve(p, (size_t)BB * HH * 4);
    unsigned int* linkW_pk = (unsigned int*)carve(p, (size_t)HH * HH * 4);
    unsigned int* genW_pk  = (unsigned int*)carve(p, (size_t)HH * HH * 4);

    // ---- stage 0: conversions / gathers / packing
    gather_cast_kernel<<<NN, 256, 0, stream>>>(in_nodes,  node_emb, (unsigned int*)xin_bf);
    gather_cast_kernel<<<NN, 256, 0, stream>>>(out_nodes, node_emb, (unsigned int*)xout_bf);
    {   // W_att: flat [NH*512][128]; head blocks even-row aligned so pairs never straddle
        int total = (NH * F_IN / 2) * DH;
        pack_b_kernel<<<(total + 255) / 256, 256, 0, stream>>>(W_att, watt_pk, NH * F_IN, DH);
    }
    {
        int total = (HH / 2) * H3;
        int g = (total + 255) / 256;
        pack_gruw_kernel<<<g, 256, 0, stream>>>(gin_Wi,  WiP_in);
        pack_gruw_kernel<<<g, 256, 0, stream>>>(gin_Wh,  WhP_in);
        pack_gruw_kernel<<<g, 256, 0, stream>>>(gout_Wi, WiP_out);
        pack_gruw_kernel<<<g, 256, 0, stream>>>(gout_Wh, WhP_out);
    }
    {
        int total = HH * HH;   // (2H/2) * H
        pack_b_kernel<<<(total + 255) / 256, 256, 0, stream>>>(link_W, linkW_pk, 2 * HH, HH);
        pack_b_kernel<<<(total + 255) / 256, 256, 0, stream>>>(gen_W,  genW_pk,  2 * HH, HH);
    }

    // ---- stage 1: GAT head projections h = x @ W  (z over heads)
    {
        dim3 grid(NN / 64, DH / 64, NH);
        gemm_bf16_kernel<<<grid, 128, 0, stream>>>(
            xin_bf, watt_pk, nullptr, hin, NN, DH, F_IN,
            0LL, (long long)(F_IN / 2) * DH, (long long)NN * DH);
        gemm_bf16_kernel<<<grid, 128, 0, stream>>>(
            xout_bf, watt_pk, nullptr, hout, NN, DH, F_IN,
            0LL, (long long)(F_IN / 2) * DH, (long long)NN * DH);
    }

    // ---- stage 2: f1/f2 + fused softmax attention rows
    {
        dim3 grid(NN, NH);
        f1f2_kernel<<<grid, 128, 0, stream>>>(hin,  a1, a2, f1in,  f2in);
        f1f2_kernel<<<grid, 128, 0, stream>>>(hout, a1, a2, f1out, f2out);
        gat_att_row_kernel<<<grid, 128, 0, stream>>>(hin,  inner_adj, f1in,  f2in,  emb_in);
        gat_att_row_kernel<<<grid, 128, 0, stream>>>(hout, outer_adj, f1out, f2out, emb_out);
    }

    // ---- stage 3: GRU text encoders (persistent over 64 timesteps)
    gru_kernel<<<BC / 16, 128, 0, stream>>>(in_ctx,  in_len,  word_emb,
                                            WiP_in,  WhP_in,  gin_bi,  gin_bh,  inh_seq);
    gru_kernel<<<BC / 16, 128, 0, stream>>>(out_ctx, out_len, word_emb,
                                            WiP_out, WhP_out, gout_bi, gout_bh, outh_seq);

    // ---- stage 4: head/tail assembly (elu graph part + ctx-avg text part)
    build_headtail_kernel<<<BB, 256, 0, stream>>>(
        emb_in, emb_out, in_map_hs, in_map_ts, out_map_hs, out_map_ts,
        inh_seq, outh_seq, headb, tailb);

    // ---- stage 5: output linears straight into d_out
    {
        dim3 grid(BB / 64, HH / 64, 1);
        const size_t seg = (size_t)BB * HH;
        gemm_bf16_kernel<<<grid, 128, 0, stream>>>((const bf16_t*)headb, linkW_pk, link_b,
                                                   out + 0 * seg, BB, HH, 2 * HH, 0, 0, 0);
        gemm_bf16_kernel<<<grid, 128, 0, stream>>>((const bf16_t*)tailb, linkW_pk, link_b,
                                                   out + 1 * seg, BB, HH, 2 * HH, 0, 0, 0);
        gemm_bf16_kernel<<<grid, 128, 0, stream>>>((const bf16_t*)headb, genW_pk,  gen_b,
                                                   out + 2 * seg, BB, HH, 2 * HH, 0, 0, 0);
        gemm_bf16_kernel<<<grid, 128, 0, stream>>>((const bf16_t*)tailb, genW_pk,  gen_b,
                                                   out + 3 * seg, BB, HH, 2 * HH, 0, 0, 0);
    }
}
